// Net_10849087390578
// MI455X (gfx1250) — compile-verified, matrix operand order checked by reading, not run on hardware
//
#include <hip/hip_runtime.h>
#include <hip/hip_bf16.h>

// ---------------------------------------------------------------------------
// RandLA-Net-like point cloud classifier for MI455X (gfx1250).
// - All dense contractions: V_WMMA_F32_16X16X32_F16 (f32 accumulate).
// - GEMM loads are unconditional (clamped addresses; only K-dim is zeroed),
//   so the 32 fragment loads pipeline with a single wait chain.
// - Branchless epilogue: every activation is leaky-relu(slope); slope=1 none,
//   0.2/0.01 leaky, 0.0 relu -> fmaxf/fminf, no exec-mask diamonds.
// - KNN stages pos tiles into LDS via the Tensor Data Mover (double-buffered
//   TENSOR_LOAD_TO_LDS + s_wait_tensorcnt), distances read from LDS.
// ---------------------------------------------------------------------------

#define BB   2
#define NN   16384
#define KNN_K 16
#define DECIM 4

typedef __attribute__((ext_vector_type(16))) _Float16 v16h;
typedef __attribute__((ext_vector_type(8)))  float    v8f;
typedef __attribute__((ext_vector_type(4)))  unsigned tdm_u4;
typedef __attribute__((ext_vector_type(8)))  int      tdm_i8;
typedef __attribute__((ext_vector_type(4)))  int      tdm_i4;

static inline unsigned cdiv(size_t a, size_t b) { return (unsigned)((a + b - 1) / b); }

__device__ __forceinline__ float leaky(float v, float slope)
{   // slope==1 -> identity, slope==0 -> relu, else leaky-relu. Branchless.
    return fmaxf(v, 0.f) + slope * fminf(v, 0.f);
}

// ======================= generic WMMA GEMM ==================================
// C[M,N] = leaky( A[M,Kd] @ W[Kd,N] + bias, slope ), wave per 16x16 tile.
__global__ void k_gemm(const float* __restrict__ A, const float* __restrict__ W,
                       const float* __restrict__ bias, float* __restrict__ C,
                       int M, int N, int Kd, float slope)
{
    int lane = threadIdx.x & 31;
    int wave = blockIdx.x * (blockDim.x >> 5) + (threadIdx.x >> 5);
    int tilesN = (N + 15) >> 4;
    int tilesM = (M + 15) >> 4;
    if (wave >= tilesM * tilesN) return;
    int tileM = wave / tilesN;
    int tileN = wave % tilesN;

    int m  = tileM * 16 + (lane & 15);
    int nn = tileN * 16 + (lane & 15);
    int khalf = (lane >> 4) * 8;

    // Clamp row/col for addressing only: OOB rows/cols feed outputs we never
    // store, so their (garbage) contents are harmless. Only k>=Kd must be 0.
    int mc = (m  < M) ? m  : (M - 1);
    int nc = (nn < N) ? nn : (N - 1);
    const float* __restrict__ Arow = A + (size_t)mc * Kd;

    v8f acc = {0.f, 0.f, 0.f, 0.f, 0.f, 0.f, 0.f, 0.f};

    for (int k0 = 0; k0 < Kd; k0 += 32) {
        if ((k0 + 32) < Kd)
            __builtin_prefetch(&Arow[k0 + 32], 0, 1); // global_prefetch_b8

        float ta[16], tb[16];
#pragma unroll
        for (int i = 0; i < 16; ++i) {          // unconditional, pipelined loads
            int g = i >> 1, h = i & 1;
            int k = k0 + ((g < 4) ? g * 2 : 16 + (g - 4) * 2) + h + khalf;
            int kc = (k < Kd) ? k : (Kd - 1);
            ta[i] = Arow[kc];
            tb[i] = W[(size_t)kc * N + nc];
        }
        v16h af, bf;
#pragma unroll
        for (int i = 0; i < 16; ++i) {          // zero-fill only the K edge
            int g = i >> 1, h = i & 1;
            int k = k0 + ((g < 4) ? g * 2 : 16 + (g - 4) * 2) + h + khalf;
            bool ok = (k < Kd);
            af[i] = (_Float16)(ok ? ta[i] : 0.f);
            bf[i] = (_Float16)(ok ? tb[i] : 0.f);
        }
        acc = __builtin_amdgcn_wmma_f32_16x16x32_f16(false, af, false, bf,
                                                     (short)0, acc, false, false);
    }

    // Hoisted, clamped bias load: one load, no per-element diamonds.
    float bval = bias ? bias[nc] : 0.f;
    int col = tileN * 16 + (lane & 15);
#pragma unroll
    for (int v = 0; v < 8; ++v) {
        int row = tileM * 16 + v + ((lane >> 4) << 3);
        if (row < M && col < N)
            C[(size_t)row * N + col] = leaky(acc[v] + bval, slope);
    }
}

// ======================= batch-norm statistics ==============================
__global__ void k_colstats(const float* __restrict__ X, int rows, int cols,
                           float* __restrict__ mean, float* __restrict__ rstd, float eps)
{
    int c = blockIdx.x;
    float s = 0.f, s2 = 0.f;
    for (int r = threadIdx.x; r < rows; r += blockDim.x) {
        float v = X[(size_t)r * cols + c];
        s += v; s2 += v * v;
    }
    __shared__ float sh[256], sh2[256];
    sh[threadIdx.x] = s; sh2[threadIdx.x] = s2;
    __syncthreads();
    for (int o = blockDim.x >> 1; o > 0; o >>= 1) {
        if ((int)threadIdx.x < o) { sh[threadIdx.x] += sh[threadIdx.x + o];
                                    sh2[threadIdx.x] += sh2[threadIdx.x + o]; }
        __syncthreads();
    }
    if (threadIdx.x == 0) {
        float mu  = sh[0] / (float)rows;
        float var = sh2[0] / (float)rows - mu * mu;
        mean[c] = mu;
        rstd[c] = rsqrtf(var + eps);
    }
}

__global__ void k_bn(const float* __restrict__ src, float* __restrict__ dst,
                     const float* __restrict__ mean, const float* __restrict__ rstd,
                     const float* __restrict__ g, const float* __restrict__ bt,
                     size_t total, int cols, float slope)
{
    size_t idx = (size_t)blockIdx.x * blockDim.x + threadIdx.x;
    if (idx >= total) return;
    int c = (int)(idx % cols);
    float v = src[idx];
    v = (v - mean[c]) * rstd[c] * g[c] + bt[c];
    dst[idx] = leaky(v, slope);
}

// ======================= Tensor Data Mover helper ===========================
// 1-D TDM transfer: tileElems elements of 4B from gaddr into LDS at ldsOff.
// HW zero-fills reads past validElems (tensor_dim0 OOB rule).
__device__ __forceinline__ void tdm_load_1d(unsigned ldsOff, const float* gaddr,
                                            unsigned validElems, unsigned tileElems)
{
    unsigned long long ga = (unsigned long long)(uintptr_t)gaddr;
    tdm_u4 g0;
    g0[0] = 1u;                                        // count=1 (valid user D#)
    g0[1] = ldsOff;                                    // lds_addr
    g0[2] = (unsigned)(ga & 0xFFFFFFFFu);              // global_addr[31:0]
    g0[3] = (unsigned)((ga >> 32) & 0x01FFFFFFu)       // global_addr[56:32]
          | 0x80000000u;                               // type=2 ("image")
    tdm_i8 g1;
    g1[0] = (int)(2u << 16);                           // wg_mask=0, data_size=4B
    g1[1] = (int)((validElems & 0xFFFFu) << 16);       // tensor_dim0[15:0]
    g1[2] = (int)((validElems >> 16) & 0xFFFFu);       // tensor_dim0[31:16], dim1=0
    g1[3] = (int)((tileElems & 0xFFFFu) << 16);        // tile_dim0
    g1[4] = 0;                                         // tile_dim1=0 (1-D), dim2=0
    g1[5] = 0; g1[6] = 0; g1[7] = 0;                   // strides unused for 1-D
    tdm_i4 gz4 = {0, 0, 0, 0};
    tdm_i8 gz8 = {0, 0, 0, 0, 0, 0, 0, 0};
    __builtin_amdgcn_tensor_load_to_lds(g0, g1, gz4, gz4, gz8, 0);
}

// ======================= KNN (TDM-staged tiles, wave-per-query) =============
#define KT 512                      // points per staged tile
__global__ void k_knn(const float* __restrict__ pos, int n, int m, int* __restrict__ nbr)
{
    __shared__ float tile[2][KT * 3];
    __shared__ float sd[8][512];
    __shared__ int   si[8][512];

    int lane = threadIdx.x & 31;
    int wgrp = threadIdx.x >> 5;
    int b = blockIdx.y;
    int q = blockIdx.x * 8 + wgrp;
    bool active = (q < m);
    const float* P = pos + (size_t)b * n * 3;

    float qx = 0.f, qy = 0.f, qz = 0.f;
    if (active) {
        qx = P[(size_t)q * DECIM * 3 + 0];
        qy = P[(size_t)q * DECIM * 3 + 1];
        qz = P[(size_t)q * DECIM * 3 + 2];
    }

    float bd[KNN_K]; int bi[KNN_K];
#pragma unroll
    for (int t = 0; t < KNN_K; ++t) { bd[t] = 3e38f; bi[t] = 0; }
    float wv = 3e38f; int wsl = 0;

    int ntiles = (n + KT - 1) / KT;
    if (wgrp == 0)
        tdm_load_1d((unsigned)(uintptr_t)&tile[0][0], P, (unsigned)n * 3, KT * 3);

    for (int t = 0; t < ntiles; ++t) {
        if (wgrp == 0)
            __builtin_amdgcn_s_wait_tensorcnt(0);   // tile t landed in LDS
        __syncthreads();
        if (wgrp == 0 && (t + 1) < ntiles) {        // prefetch tile t+1
            int s2 = (t + 1) * KT;
            tdm_load_1d((unsigned)(uintptr_t)&tile[(t + 1) & 1][0],
                        P + (size_t)s2 * 3, (unsigned)(n - s2) * 3, KT * 3);
        }
        const float* tb = tile[t & 1];
        int base = t * KT;
        int cnt = n - base; if (cnt > KT) cnt = KT;
        if (active) {
            for (int j = lane; j < cnt; j += 32) {
                float dx = tb[j * 3 + 0] - qx;
                float dy = tb[j * 3 + 1] - qy;
                float dz = tb[j * 3 + 2] - qz;
                float d2 = dx * dx + dy * dy + dz * dz;
                if (d2 < wv) {
                    int gj = base + j;
#pragma unroll
                    for (int t2 = 0; t2 < KNN_K; ++t2)
                        if (t2 == wsl) { bd[t2] = d2; bi[t2] = gj; }
                    wv = bd[0]; wsl = 0;
#pragma unroll
                    for (int t2 = 1; t2 < KNN_K; ++t2)
                        if (bd[t2] > wv) { wv = bd[t2]; wsl = t2; }
                }
            }
        }
        __syncthreads();                            // done reading tile[t&1]
    }

    // ---- merge 32 lanes x 16 candidates -> best 16 ----
#pragma unroll
    for (int t = 0; t < KNN_K; ++t) {
        sd[wgrp][lane * KNN_K + t] = bd[t];
        si[wgrp][lane * KNN_K + t] = bi[t];
    }
    __syncthreads();

    for (int s = 0; s < KNN_K; ++s) {
        float best = 3e37f; int bpos = -1;
        for (int j = lane; j < 512; j += 32) {
            float v = sd[wgrp][j];
            if (v < best) { best = v; bpos = j; }
        }
        for (int off = 16; off > 0; off >>= 1) {
            float ov = __shfl_xor(best, off, 32);
            int   op = __shfl_xor(bpos, off, 32);
            if (ov < best || (ov == best && op >= 0 && op < bpos)) { best = ov; bpos = op; }
        }
        if (lane == 0 && bpos >= 0) {
            if (active) nbr[((size_t)b * m + q) * KNN_K + s] = si[wgrp][bpos];
            sd[wgrp][bpos] = 3e38f;
        }
        __syncthreads();
    }
}

// ======================= LFA helper kernels =================================
__global__ void k_rel(const float* __restrict__ pos, const int* __restrict__ nbr,
                      int n, int m, float* __restrict__ rel)
{
    size_t idx = (size_t)blockIdx.x * blockDim.x + threadIdx.x;
    size_t total = (size_t)BB * m * KNN_K;
    if (idx >= total) return;
    int b = (int)(idx / ((size_t)m * KNN_K));
    int r = (int)(idx % ((size_t)m * KNN_K));
    int i = r / KNN_K;
    const float* P = pos + (size_t)b * n * 3;
    int j = nbr[idx];
    float pix = P[(size_t)i * 3 + 0], piy = P[(size_t)i * 3 + 1], piz = P[(size_t)i * 3 + 2];
    float pjx = P[(size_t)j * 3 + 0], pjy = P[(size_t)j * 3 + 1], pjz = P[(size_t)j * 3 + 2];
    float dx = pjx - pix, dy = pjy - piy, dz = pjz - piz;
    float eu = sqrtf(dx * dx + dy * dy + dz * dz);
    float* o = rel + idx * 10;
    o[0] = pix; o[1] = piy; o[2] = piz;
    o[3] = pjx; o[4] = pjy; o[5] = pjz;
    o[6] = dx;  o[7] = dy;  o[8] = dz;  o[9] = eu;
}

__global__ void k_concat(const float* __restrict__ x, const int* __restrict__ nbr,
                         const float* __restrict__ enc, float* __restrict__ lf,
                         int n, int m, int dx, int de)
{
    int d = dx + de;
    size_t total = (size_t)BB * m * KNN_K * d;
    size_t idx = (size_t)blockIdx.x * blockDim.x + threadIdx.x;
    if (idx >= total) return;
    size_t row = idx / d;
    int c = (int)(idx % d);
    int b = (int)(row / ((size_t)m * KNN_K));
    if (c < dx) {
        int j = nbr[row];
        lf[idx] = x[((size_t)b * n + j) * dx + c];
    } else {
        lf[idx] = enc[row * de + (c - dx)];
    }
}

__global__ void k_softmax(float* __restrict__ a, size_t rows, int d)
{
    size_t r = (size_t)blockIdx.x * blockDim.x + threadIdx.x;
    if (r >= rows) return;
    float* p = a + r * d;
    float mx = -3e38f;
    for (int c = 0; c < d; ++c) mx = fmaxf(mx, p[c]);
    float s = 0.f;
    for (int c = 0; c < d; ++c) { float e = expf(p[c] - mx); p[c] = e; s += e; }
    float inv = 1.f / s;
    for (int c = 0; c < d; ++c) p[c] *= inv;
}

__global__ void k_agg(const float* __restrict__ att, const float* __restrict__ lf,
                      float* __restrict__ agg, int n, int m, int d)
{
    size_t total = (size_t)BB * m * d;
    size_t idx = (size_t)blockIdx.x * blockDim.x + threadIdx.x;
    if (idx >= total) return;
    int b = (int)(idx / ((size_t)m * d));
    int r = (int)(idx % ((size_t)m * d));
    int i = r / d, c = r % d;
    size_t base = (((size_t)b * m + i) * KNN_K) * d + c;
    float s = 0.f;
#pragma unroll
    for (int k = 0; k < KNN_K; ++k)
        s += att[base + (size_t)k * d] * lf[base + (size_t)k * d];
    agg[((size_t)b * n + i) * d + c] = s;
}

// ======================= misc elementwise ===================================
__global__ void k_add_lrelu(const float* __restrict__ a, const float* __restrict__ b,
                            float* __restrict__ o, size_t total, float slope)
{
    size_t idx = (size_t)blockIdx.x * blockDim.x + threadIdx.x;
    if (idx >= total) return;
    o[idx] = leaky(a[idx] + b[idx], slope);
}

__global__ void k_down(const float* __restrict__ src, float* __restrict__ dst,
                       int n, int m, int d)
{
    size_t total = (size_t)BB * m * d;
    size_t idx = (size_t)blockIdx.x * blockDim.x + threadIdx.x;
    if (idx >= total) return;
    int b = (int)(idx / ((size_t)m * d));
    int r = (int)(idx % ((size_t)m * d));
    int i = r / d, c = r % d;
    dst[idx] = src[((size_t)b * n + (size_t)i * DECIM) * d + c];
}

__global__ void k_maxpool(const float* __restrict__ in, float* __restrict__ out,
                          int n, int cols)
{
    int idx = blockIdx.x * blockDim.x + threadIdx.x;
    if (idx >= BB * cols) return;
    int b = idx / cols, c = idx % cols;
    float mx = -3e38f;
    for (int i = 0; i < n; ++i)
        mx = fmaxf(mx, in[((size_t)b * n + i) * cols + c]);
    out[idx] = mx;
}

__global__ void k_logsoftmax(const float* __restrict__ in, float* __restrict__ out,
                             int rows, int C)
{
    int r = threadIdx.x;
    if (r >= rows) return;
    const float* p = in + (size_t)r * C;
    float mx = -3e38f;
    for (int c = 0; c < C; ++c) mx = fmaxf(mx, p[c]);
    float s = 0.f;
    for (int c = 0; c < C; ++c) s += expf(p[c] - mx);
    float l = logf(s);
    for (int c = 0; c < C; ++c) out[(size_t)r * C + c] = p[c] - mx - l;
}

// ======================= host-side orchestration ============================
struct Lin { const float* W; const float* b; };
struct BnP { const float* g; const float* bt; };
struct LfaP { Lin enc; BnP encbn; const float* attW; Lin post; BnP postbn; };
struct BlkP { Lin mlp1, shortl; BnP shortbn; LfaP l1, l2; Lin mlp2; BnP mlp2bn; };

struct Bump {
    char* base; size_t off; size_t cap;
    float* f(size_t n) { size_t b = ((n * 4) + 255) & ~(size_t)255;
                         float* p = (float*)(base + off); off += b; return p; }
    int*   i(size_t n) { size_t b = ((n * 4) + 255) & ~(size_t)255;
                         int* p = (int*)(base + off); off += b; return p; }
};

static void gemm(hipStream_t st, const float* A, const float* W, const float* bias,
                 float* C, int M, int N, int Kd, float slope)
{
    long waves = (long)((M + 15) / 16) * ((N + 15) / 16);
    k_gemm<<<cdiv((size_t)waves, 8), 256, 0, st>>>(A, W, bias, C, M, N, Kd, slope);
}

static void stats_bn(hipStream_t st, const float* src, float* dst, int rows, int cols,
                     const BnP& p, float* mean, float* rstd, float slope)
{
    k_colstats<<<cols, 256, 0, st>>>(src, rows, cols, mean, rstd, 1e-6f);
    size_t total = (size_t)rows * cols;
    k_bn<<<cdiv(total, 256), 256, 0, st>>>(src, dst, mean, rstd, p.g, p.bt,
                                           total, cols, slope);
}

static void run_lfa(hipStream_t st, Bump& ws, const float* xin, int dx,
                    const float* pos, int n, int m, const int* nbr,
                    const LfaP& p, int d, float* hout, float* mean, float* rstd)
{
    int de = d / 2;                           // == dx
    size_t R = (size_t)BB * m * KNN_K;
    float* rel  = ws.f(R * 10);
    float* encb = ws.f(R * de);
    float* lf   = ws.f(R * d);
    float* att  = ws.f(R * d);
    float* agg  = ws.f((size_t)BB * n * d);
    float* pst  = ws.f((size_t)BB * n * d);

    k_rel<<<cdiv(R, 256), 256, 0, st>>>(pos, nbr, n, m, rel);
    gemm(st, rel, p.enc.W, p.enc.b, encb, (int)R, de, 10, 1.0f);
    stats_bn(st, encb, encb, (int)R, de, p.encbn, mean, rstd, 0.2f);
    k_concat<<<cdiv(R * d, 256), 256, 0, st>>>(xin, nbr, encb, lf, n, m, dx, de);
    gemm(st, lf, p.attW, nullptr, att, (int)R, d, d, 1.0f);
    k_softmax<<<cdiv(R, 256), 256, 0, st>>>(att, R, d);
    (void)hipMemsetAsync(agg, 0, (size_t)BB * n * d * 4, st);
    k_agg<<<cdiv((size_t)BB * m * d, 256), 256, 0, st>>>(att, lf, agg, n, m, d);
    gemm(st, agg, p.post.W, p.post.b, pst, BB * n, d, d, 1.0f);
    stats_bn(st, pst, hout, BB * n, d, p.postbn, mean, rstd, 0.2f);
}

static void run_block(hipStream_t st, Bump& ws, const float* xin, int din,
                      const float* posin, int n, int dout, const BlkP& p,
                      float* hnext, float* posnext, float* mean, float* rstd)
{
    int m = n / DECIM;
    int d8 = dout / 8, d1 = dout / 4, d2 = dout / 2;
    int* nbr    = ws.i((size_t)BB * m * KNN_K);
    float* sc   = ws.f((size_t)BB * n * dout);
    float* h1   = ws.f((size_t)BB * n * d8);
    float* h2   = ws.f((size_t)BB * n * d1);
    float* h3   = ws.f((size_t)BB * n * d2);
    float* m2   = ws.f((size_t)BB * n * dout);
    float* hful = ws.f((size_t)BB * n * dout);
    size_t lfaMark = ws.off;

    k_knn<<<dim3(cdiv(m, 8), BB), 256, 0, st>>>(posin, n, m, nbr);
    gemm(st, xin, p.shortl.W, p.shortl.b, sc, BB * n, dout, din, 1.0f);
    stats_bn(st, sc, sc, BB * n, dout, p.shortbn, mean, rstd, 1.0f);
    gemm(st, xin, p.mlp1.W, p.mlp1.b, h1, BB * n, d8, din, 0.2f);

    run_lfa(st, ws, h1, d8, posin, n, m, nbr, p.l1, d1, h2, mean, rstd);
    ws.off = lfaMark;
    run_lfa(st, ws, h2, d1, posin, n, m, nbr, p.l2, d2, h3, mean, rstd);
    ws.off = lfaMark;

    gemm(st, h3, p.mlp2.W, p.mlp2.b, m2, BB * n, dout, d2, 1.0f);
    stats_bn(st, m2, m2, BB * n, dout, p.mlp2bn, mean, rstd, 1.0f);
    k_add_lrelu<<<cdiv((size_t)BB * n * dout, 256), 256, 0, st>>>(
        m2, sc, hful, (size_t)BB * n * dout, 0.01f);
    k_down<<<cdiv((size_t)BB * m * dout, 256), 256, 0, st>>>(hful, hnext, n, m, dout);
    k_down<<<cdiv((size_t)BB * m * 3, 256), 256, 0, st>>>(posin, posnext, n, m, 3);
}

extern "C" void kernel_launch(void* const* d_in, const int* in_sizes, int n_in,
                              void* d_out, int out_size, void* d_ws, size_t ws_size,
                              hipStream_t stream)
{
    (void)in_sizes; (void)n_in; (void)out_size;

    int pi = 0;
    auto NXT = [&]() { return (const float*)d_in[pi++]; };
    const float* x    = NXT();
    const float* pos0 = NXT();
    auto lin = [&](bool bias) { Lin l; l.W = NXT(); l.b = bias ? NXT() : nullptr; return l; };
    auto bnp = [&]() { BnP b; b.g = NXT(); b.bt = NXT(); return b; };
    auto lfp = [&]() { LfaP f; f.enc = lin(true); f.encbn = bnp();
                       f.attW = NXT(); f.post = lin(true); f.postbn = bnp(); return f; };
    auto blk = [&]() { BlkP b; b.mlp1 = lin(true); b.shortl = lin(true); b.shortbn = bnp();
                       b.l1 = lfp(); b.l2 = lfp(); b.mlp2 = lin(true); b.mlp2bn = bnp();
                       return b; };
    Lin fc0 = lin(true); BnP fc0bn = bnp();
    BlkP b1 = blk(), b2 = blk(), b3 = blk(), b4 = blk();
    Lin pool = lin(true); BnP poolbn = bnp();
    Lin head1 = lin(true); BnP h1bn = bnp();
    Lin head2 = lin(true);

    Bump ws{(char*)d_ws, 0, ws_size};
    float* mean = ws.f(512);
    float* rstd = ws.f(512);
    float* h0   = ws.f((size_t)BB * NN * 8);
    float* Hp[2] = { ws.f(262144), ws.f(262144) };
    float* Pp[2] = { ws.f(24576),  ws.f(24576)  };
    size_t scratch0 = ws.off;

    gemm(stream, x, fc0.W, fc0.b, h0, BB * NN, 8, 3, 1.0f);
    stats_bn(stream, h0, h0, BB * NN, 8, fc0bn, mean, rstd, 1.0f);

    const BlkP* blks[4] = { &b1, &b2, &b3, &b4 };
    int douts[4] = { 32, 128, 256, 512 };
    const float* hc = h0; const float* pc = pos0;
    int nc = NN, dc = 8;
    for (int i = 0; i < 4; ++i) {
        ws.off = scratch0;
        run_block(stream, ws, hc, dc, pc, nc, douts[i], *blks[i],
                  Hp[i & 1], Pp[i & 1], mean, rstd);
        hc = Hp[i & 1]; pc = Pp[i & 1];
        nc /= DECIM; dc = douts[i];
    }

    ws.off = scratch0;
    int np = nc;                              // 64
    float* t1 = ws.f((size_t)BB * np * 512);
    gemm(stream, hc, pool.W, pool.b, t1, BB * np, 512, 512, 1.0f);
    stats_bn(stream, t1, t1, BB * np, 512, poolbn, mean, rstd, 0.0f);
    float* hv = ws.f((size_t)BB * 512);
    k_maxpool<<<cdiv((size_t)BB * 512, 256), 256, 0, stream>>>(t1, hv, np, 512);
    float* t2 = ws.f((size_t)BB * 64);
    gemm(stream, hv, head1.W, head1.b, t2, BB, 64, 512, 1.0f);
    stats_bn(stream, t2, t2, BB, 64, h1bn, mean, rstd, 0.0f);
    float* t3 = ws.f((size_t)BB * 40);
    gemm(stream, t2, head2.W, head2.b, t3, BB, 40, 64, 1.0f);
    k_logsoftmax<<<1, 32, 0, stream>>>(t3, (float*)d_out, BB, 40);
}
